// SingleAttention_44813688767090
// MI455X (gfx1250) — matrix operand order repaired and checked by reading, not compiled, not run on hardware
//
#include <hip/hip_runtime.h>
#include <hip/hip_bf16.h>

// ---------------------------------------------------------------------------
// Fused single-head attention for MI455X (gfx1250, wave32, WMMA + TDM).
//   B=4, S=2048, D=512, DK=DV=64
// Kernel 1: QKV projection fp32->f16 GEMMs via v_wmma_f32_16x16x32_f16,
//           Q pre-scaled by (1/sqrt(64))*log2(e); V stored transposed.
// Kernel 2: flash attention (online softmax in log2 domain); K/V tiles are
//           double-buffered in LDS via the Tensor Data Mover
//           (tensor_load_to_lds + s_wait_tensorcnt), WMMA for QK^T and PV.
// ---------------------------------------------------------------------------

typedef __attribute__((ext_vector_type(16))) _Float16 v16h;
typedef __attribute__((ext_vector_type(8)))  _Float16 v8h;
typedef __attribute__((ext_vector_type(4)))  _Float16 v4h;
typedef __attribute__((ext_vector_type(8)))  float    v8f;
typedef __attribute__((ext_vector_type(4)))  float    v4f;

#define NB   4
#define SEQ  2048
#define DM   512
#define DKV  64

#if __has_builtin(__builtin_amdgcn_tensor_load_to_lds)
#define HAS_TDM 1
#else
#define HAS_TDM 0
#endif

__device__ __forceinline__ v16h cat8(v8h lo, v8h hi) {
    return __builtin_shufflevector(lo, hi, 0,1,2,3,4,5,6,7,8,9,10,11,12,13,14,15);
}

__device__ __forceinline__ v8f wmma_f16(v16h a, v16h b, v8f c) {
    return __builtin_amdgcn_wmma_f32_16x16x32_f16(false, a, false, b,
                                                  (short)0, c, false, false);
}

__device__ __forceinline__ float rmax16(float x) {
    x = fmaxf(x, __shfl_xor(x, 1));
    x = fmaxf(x, __shfl_xor(x, 2));
    x = fmaxf(x, __shfl_xor(x, 4));
    x = fmaxf(x, __shfl_xor(x, 8));
    return x;
}
__device__ __forceinline__ float rsum16(float x) {
    x += __shfl_xor(x, 1);
    x += __shfl_xor(x, 2);
    x += __shfl_xor(x, 4);
    x += __shfl_xor(x, 8);
    return x;
}

#if HAS_TDM
typedef __attribute__((ext_vector_type(4))) unsigned int u32x4;
typedef __attribute__((ext_vector_type(8))) int          i32x8;
typedef __attribute__((ext_vector_type(4))) int          i32x4;

// Build a 2D TDM descriptor (D# groups 0/1 per CDNA5 ISA ch.8) and issue
// TENSOR_LOAD_TO_LDS.  Tile = tile_d0 x tile_d1 elements of 2-byte data,
// innermost dimension stride = stride0 elements.
__device__ __forceinline__ void tdm_load_2d(const _Float16* gptr, unsigned lds_off,
                                            unsigned tensor_d0, unsigned tensor_d1,
                                            unsigned tile_d0, unsigned tile_d1,
                                            unsigned stride0)
{
    unsigned long long ga = (unsigned long long)(uintptr_t)gptr;
    u32x4 g0;
    g0[0] = 1u;                                    // count=1 (valid), no gather
    g0[1] = lds_off;                               // lds_addr (bytes)
    g0[2] = (unsigned)(ga & 0xffffffffu);          // global_addr[31:0]
    g0[3] = (unsigned)((ga >> 32) & 0x01ffffffu)   // global_addr[56:32]
          | 0x80000000u;                           // type = 2 ("image")
    i32x8 g1;
    g1[0] = 0x00010000;                            // data_size=1 (2 bytes), mask=0
    g1[1] = (int)((tensor_d0 & 0xffffu) << 16);    // tensor_dim0[15:0] @ [63:48]
    g1[2] = (int)(((tensor_d0 >> 16) & 0xffffu)    // tensor_dim0[31:16]
          | ((tensor_d1 & 0xffffu) << 16));        // tensor_dim1[15:0]
    g1[3] = (int)(((tensor_d1 >> 16) & 0xffffu)    // tensor_dim1[31:16]
          | (tile_d0 << 16));                      // tile_dim0
    g1[4] = (int)(tile_d1 & 0xffffu);              // tile_dim1 (tile_dim2 = 0)
    g1[5] = (int)stride0;                          // tensor_dim0_stride[31:0]
    g1[6] = 0;                                     // stride0[47:32] / dim1_stride
    g1[7] = 0;
    i32x4 z4 = {0, 0, 0, 0};
    i32x8 z8 = {0, 0, 0, 0, 0, 0, 0, 0};
    // 6-arg toolchain form: (g0, g1, g2, g3, g_ext, cpol)
    __builtin_amdgcn_tensor_load_to_lds(g0, g1, z4, z4, z8, 0);
}
#endif

// ---------------------------------------------------------------------------
// Kernel 1: projection GEMM  out[m][n] = sum_k X[m][k] * W[n][k]
// grid.x = 3 * 64 ; block = 256 (8 waves); 128 rows per block.
// mat 0 -> Qh (scaled), mat 1 -> Kh, mat 2 -> Vt (transposed [B,64,S])
// ---------------------------------------------------------------------------
__global__ __launch_bounds__(256)
void qkv_proj_kernel(const float* __restrict__ xq, const float* __restrict__ xk,
                     const float* __restrict__ xv,
                     const float* __restrict__ Wq, const float* __restrict__ Wk,
                     const float* __restrict__ Wv,
                     _Float16* __restrict__ Qh, _Float16* __restrict__ Kh,
                     _Float16* __restrict__ Vt)
{
    const int mat  = blockIdx.x >> 6;          // 0=Q,1=K,2=V
    const int tile = blockIdx.x & 63;
    const int row0 = tile * 128;

    const float* __restrict__ X = (mat == 0) ? xq : (mat == 1) ? xk : xv;
    const float* __restrict__ W = (mat == 0) ? Wq : (mat == 1) ? Wk : Wv;

    __shared__ __attribute__((aligned(32))) _Float16 sX[128 * 32];
    __shared__ __attribute__((aligned(32))) _Float16 sW[64 * 32];

    const int tid  = threadIdx.x;
    const int lane = tid & 31;
    const int wave = tid >> 5;
    const int l16  = lane & 15;
    const int hi   = lane >> 4;          // half-wave select
    const int mrow = wave * 16 + l16;    // local A row for this lane

    v8f acc[4] = {v8f{}, v8f{}, v8f{}, v8f{}};

    for (int k0 = 0; k0 < DM; k0 += 32) {
        __syncthreads();
        // ---- stage X tile [128 x 32] fp32 -> f16 in LDS (4 float4 / thread)
        #pragma unroll
        for (int i = 0; i < 4; ++i) {
            int slot = tid + i * 256;             // 0..1023
            int r = slot >> 3, c = (slot & 7) * 4;
            v4f t = *(const v4f*)(X + (size_t)(row0 + r) * DM + k0 + c);
            v4h h; h[0] = (_Float16)t.x; h[1] = (_Float16)t.y;
                   h[2] = (_Float16)t.z; h[3] = (_Float16)t.w;
            *(v4h*)&sX[r * 32 + c] = h;
        }
        // ---- stage W tile [64 x 32] fp32 -> f16 (2 float4 / thread)
        #pragma unroll
        for (int i = 0; i < 2; ++i) {
            int slot = tid + i * 256;             // 0..511
            int r = slot >> 3, c = (slot & 7) * 4;
            v4f t = *(const v4f*)(W + (size_t)r * DM + k0 + c);
            v4h h; h[0] = (_Float16)t.x; h[1] = (_Float16)t.y;
                   h[2] = (_Float16)t.z; h[3] = (_Float16)t.w;
            *(v4h*)&sW[r * 32 + c] = h;
        }
        __syncthreads();

        // ---- A fragment (16x32 f16 layout: K 0..7/16..23 lanes0-15, +8 hi)
        v8h alo = *(const v8h*)&sX[mrow * 32 + hi * 8];
        v8h ahi = *(const v8h*)&sX[mrow * 32 + 16 + hi * 8];
        v16h a  = cat8(alo, ahi);

        // ---- B fragments: same K order as A (0..7/16..23, +8 hi)
        #pragma unroll
        for (int nt = 0; nt < 4; ++nt) {
            v16h b = cat8(*(const v8h*)&sW[(nt * 16 + l16) * 32 + hi * 8],
                          *(const v8h*)&sW[(nt * 16 + l16) * 32 + 16 + hi * 8]);
            acc[nt] = wmma_f16(a, b, acc[nt]);
        }
    }

    // fold 1/sqrt(dk) * log2(e) into Q so softmax runs in exp2 domain
    const float scale = (mat == 0) ? 0.18033688011112042f : 1.0f;

    #pragma unroll
    for (int nt = 0; nt < 4; ++nt) {
        #pragma unroll
        for (int r = 0; r < 8; ++r) {
            _Float16 h = (_Float16)(acc[nt][r] * scale);
            int grow = row0 + wave * 16 + r + hi * 8;   // 0..8191
            int col  = nt * 16 + l16;
            if (mat == 2) {
                int b = grow >> 11, s = grow & (SEQ - 1);
                Vt[((size_t)b * DKV + col) * SEQ + s] = h;
            } else {
                _Float16* O = (mat == 0) ? Qh : Kh;
                O[(size_t)grow * DKV + col] = h;
            }
        }
    }
}

// ---------------------------------------------------------------------------
// Kernel 2: flash attention with TDM double-buffered K/V tiles.
// grid.x = B * (S/64) = 128 ; block = 128 (4 waves, 16 q-rows each).
// ---------------------------------------------------------------------------
__global__ __launch_bounds__(128)
void flash_attn_kernel(const _Float16* __restrict__ Qh,
                       const _Float16* __restrict__ Kh,
                       const _Float16* __restrict__ Vt,
                       float* __restrict__ out)
{
    const int b    = blockIdx.x >> 5;
    const int row0 = (blockIdx.x & 31) * 64;
    const int tid  = threadIdx.x;
    const int lane = tid & 31;
    const int wave = tid >> 5;
    const int l16  = lane & 15;
    const int hi   = lane >> 4;

    __shared__ __attribute__((aligned(32))) _Float16 sQ[64 * DKV];       // 8 KB
    __shared__ __attribute__((aligned(32))) _Float16 sP[4][16 * 32];     // 4 KB
    __shared__ __attribute__((aligned(32))) _Float16 sK[2][32 * DKV];    // 8 KB
    __shared__ __attribute__((aligned(32))) _Float16 sV[2][DKV * 32];    // 8 KB

    // ---- stage the 64x64 f16 Q tile in LDS (each thread: 32 bytes)
    {
        const _Float16* src = Qh + ((size_t)b * SEQ + row0 + (tid >> 1)) * DKV
                                 + (tid & 1) * 32;
        *(v16h*)&sQ[(tid >> 1) * DKV + (tid & 1) * 32] = *(const v16h*)src;
    }
    __syncthreads();

    // ---- per-wave Q A-fragments (loaded once, reused for all key tiles)
    const int qrow = wave * 16 + l16;
    v16h aQ[2];
    #pragma unroll
    for (int ks = 0; ks < 2; ++ks) {
        v8h lo = *(const v8h*)&sQ[qrow * DKV + ks * 32 + hi * 8];
        v8h hh = *(const v8h*)&sQ[qrow * DKV + ks * 32 + 16 + hi * 8];
        aQ[ks] = cat8(lo, hh);
    }

    const _Float16* __restrict__ Kb = Kh + (size_t)b * SEQ * DKV;
    const _Float16* __restrict__ Vb = Vt + (size_t)b * DKV * SEQ;

    v8f o[4] = {v8f{}, v8f{}, v8f{}, v8f{}};
    float mrun[8], lrun[8];
    #pragma unroll
    for (int r = 0; r < 8; ++r) { mrun[r] = -3.0e38f; lrun[r] = 0.0f; }

#if HAS_TDM
    const unsigned kds[2] = { (unsigned)(uintptr_t)&sK[0][0],
                              (unsigned)(uintptr_t)&sK[1][0] };
    const unsigned vds[2] = { (unsigned)(uintptr_t)&sV[0][0],
                              (unsigned)(uintptr_t)&sV[1][0] };
    if (wave == 0) {                       // prime the pipeline: tile 0 -> buf 0
        tdm_load_2d(Kb, kds[0], DKV, SEQ, DKV, 32, DKV);   // 32 keys x 64 feat
        tdm_load_2d(Vb, vds[0], SEQ, DKV, 32, DKV, SEQ);   // 64 cols x 32 keys
    }
#endif

    for (int jt = 0; jt < SEQ / 32; ++jt) {
        const int j0  = jt * 32;
        const int buf = jt & 1;

#if HAS_TDM
        if (wave == 0) __builtin_amdgcn_s_wait_tensorcnt(0);  // tile jt landed
        __syncthreads();
        if (wave == 0 && jt + 1 < SEQ / 32) {                 // stage tile jt+1
            const int jn = j0 + 32;
            tdm_load_2d(Kb + (size_t)jn * DKV, kds[buf ^ 1], DKV, SEQ, DKV, 32, DKV);
            tdm_load_2d(Vb + jn,               vds[buf ^ 1], SEQ, DKV, 32, DKV, SEQ);
        }
#else
        __syncthreads();
        // synchronous fallback staging: 4 KB K + 4 KB V, 32B per thread each
        *(v16h*)&sK[buf][tid * 16] =
            *(const v16h*)(Kb + (size_t)j0 * DKV + tid * 16);
        {
            int n = tid >> 1, p = tid & 1;
            *(v16h*)&sV[buf][n * 32 + p * 16] =
                *(const v16h*)(Vb + (size_t)n * SEQ + j0 + p * 16);
        }
        __syncthreads();
#endif

        // ---- scores: two 16-key subtiles, K-dim 64 = 2 WMMA steps each
        v8f sA = {}, sB = {};
        #pragma unroll
        for (int ks = 0; ks < 2; ++ks) {
            v16h bKA = cat8(*(const v8h*)&sK[buf][(l16)      * DKV + ks * 32 + hi * 8],
                            *(const v8h*)&sK[buf][(l16)      * DKV + ks * 32 + 16 + hi * 8]);
            v16h bKB = cat8(*(const v8h*)&sK[buf][(16 + l16) * DKV + ks * 32 + hi * 8],
                            *(const v8h*)&sK[buf][(16 + l16) * DKV + ks * 32 + 16 + hi * 8]);
            sA = wmma_f16(aQ[ks], bKA, sA);
            sB = wmma_f16(aQ[ks], bKB, sB);
        }

        // ---- online softmax (log2 domain; scale folded into Q)
        #pragma unroll
        for (int r = 0; r < 8; ++r) {
            float tmax = rmax16(fmaxf(sA[r], sB[r]));
            float nm   = fmaxf(mrun[r], tmax);
            float corr = exp2f(mrun[r] - nm);
            float pa   = exp2f(sA[r] - nm);
            float pb   = exp2f(sB[r] - nm);
            lrun[r] = lrun[r] * corr + rsum16(pa + pb);
            mrun[r] = nm;
            #pragma unroll
            for (int nt = 0; nt < 4; ++nt) o[nt][r] *= corr;
            // scatter P (rows r+hi*8, cols l16 / 16+l16) into wave-local LDS
            sP[wave][(r + hi * 8) * 32 + l16]      = (_Float16)pa;
            sP[wave][(r + hi * 8) * 32 + 16 + l16] = (_Float16)pb;
        }
        asm volatile("s_wait_dscnt 0x0" ::: "memory");

        // ---- transpose P out of LDS into A-fragment layout
        v8h plo = *(const v8h*)&sP[wave][l16 * 32 + hi * 8];
        v8h phi = *(const v8h*)&sP[wave][l16 * 32 + 16 + hi * 8];
        v16h aP = cat8(plo, phi);

        // ---- O += P * V  (V transposed -> B fragments in A's K order)
        #pragma unroll
        for (int nt = 0; nt < 4; ++nt) {
            v16h bV = cat8(*(const v8h*)&sV[buf][(nt * 16 + l16) * 32 + hi * 8],
                           *(const v8h*)&sV[buf][(nt * 16 + l16) * 32 + 16 + hi * 8]);
            o[nt] = wmma_f16(aP, bV, o[nt]);
        }
    }

    // ---- epilogue: normalize and store fp32
    #pragma unroll
    for (int nt = 0; nt < 4; ++nt) {
        #pragma unroll
        for (int r = 0; r < 8; ++r) {
            int m   = row0 + wave * 16 + r + hi * 8;
            int col = nt * 16 + l16;
            out[((size_t)b * SEQ + m) * DKV + col] = o[nt][r] / lrun[r];
        }
    }
}

// ---------------------------------------------------------------------------
extern "C" void kernel_launch(void* const* d_in, const int* in_sizes, int n_in,
                              void* d_out, int out_size, void* d_ws, size_t ws_size,
                              hipStream_t stream) {
    const float* xq = (const float*)d_in[0];
    const float* xk = (const float*)d_in[1];
    const float* xv = (const float*)d_in[2];
    const float* Wq = (const float*)d_in[3];
    const float* Wk = (const float*)d_in[4];
    const float* Wv = (const float*)d_in[5];

    // workspace: Q | K | Vt, each B*S*64 f16 = 1 MiB
    _Float16* Qh = (_Float16*)d_ws;
    _Float16* Kh = (_Float16*)((char*)d_ws + (1u << 20));
    _Float16* Vt = (_Float16*)((char*)d_ws + (2u << 20));

    qkv_proj_kernel<<<dim3(3 * 64), dim3(256), 0, stream>>>(
        xq, xk, xv, Wq, Wk, Wv, Qh, Kh, Vt);
    flash_attn_kernel<<<dim3(NB * (SEQ / 64)), dim3(128), 0, stream>>>(
        Qh, Kh, Vt, (float*)d_out);
}
